// Encoder_83897891160869
// MI455X (gfx1250) — compile-verified, matrix operand order checked by reading, not run on hardware
//
#include <hip/hip_runtime.h>
#include <hip/hip_bf16.h>

// ---------------------------------------------------------------------------
// 2-layer bidirectional LSTM for MI455X (gfx1250, wave32, WMMA).
// B=32, T=2048, E=256, H=256.
//
// Pipeline per layer:
//   1) proj_kernel : XZ = X(f16) @ W^T(f16) + b   (parallel WMMA GEMM, f32 acc)
//   2) scan_kernel : recurrence is independent per batch row -> split batch
//      into two 16-row halves: 4 persistent WGs per layer (dir x batch-half),
//      each 16 waves on its own WGP, no cross-WG sync. Per step:
//      z = h @ U via v_wmma_f32_16x16x32_f16 (wave owns matching i/f/g/o
//      tiles), gate math in registers, cell state c in VGPRs for all T steps,
//      h shared through padded LDS (conflict-free ds_read_b128).
// ---------------------------------------------------------------------------

typedef __attribute__((ext_vector_type(16))) _Float16 v16h;
typedef __attribute__((ext_vector_type(8)))  _Float16 v8h;
typedef __attribute__((ext_vector_type(8)))  float    v8f;

#define BB 32
#define TT 2048
#define EE 256
#define HH 256
#define N4H 1024          // 4*H
#define HROW 264          // LDS row stride in f16 (256 + 8 pad -> conflict-free b128 reads)

// Load one 16x32 f16 WMMA fragment (A-pattern; B uses the identical pattern on
// the transposed weight matrix). Lane L: row = L&15, K-half = (L<16 ? 0 : 8).
// Two 16-byte loads: K = kb..kb+7 and K = kb+16..kb+23.
__device__ __forceinline__ v16h ldfrag(const _Float16* p) {
  v8h lo = *(const v8h*)p;
  v8h hi = *(const v8h*)(p + 16);
  v16h r;
#pragma unroll
  for (int i = 0; i < 8; ++i) { r[i] = lo[i]; r[i + 8] = hi[i]; }
  return r;
}

__device__ __forceinline__ float sigmoidf_(float x) {
  return 1.0f / (1.0f + __expf(-x));
}

// ------------------------- weight prep kernels ------------------------------

// dst[n*K + k] = (f16) src[k*N + n]   (transpose + f32->f16)
__global__ void transpose_cvt_kernel(const float* __restrict__ src,
                                     _Float16* __restrict__ dst, int K, int N) {
  int idx = blockIdx.x * blockDim.x + threadIdx.x;
  if (idx >= K * N) return;
  int n = idx / K, k = idx % K;
  dst[idx] = (_Float16)src[k * N + n];
}

__global__ void cvt_f16_kernel(const float* __restrict__ src,
                               _Float16* __restrict__ dst, int n) {
  int idx = blockIdx.x * blockDim.x + threadIdx.x;
  if (idx < n) dst[idx] = (_Float16)src[idx];
}

// ------------------------- input projection GEMM ----------------------------
// C[M,1024] = A[M,K](f16) @ WT[1024,K]^T(f16) + bias. One wave = 16x64 strip.
__global__ __launch_bounds__(256) void proj_kernel(
    const _Float16* __restrict__ A, const _Float16* __restrict__ WT,
    const float* __restrict__ bias, float* __restrict__ C, int M, int K) {
  const int gw   = (blockIdx.x * blockDim.x + threadIdx.x) >> 5;
  const int lane = threadIdx.x & 31;
  const int mt = gw >> 4;          // 1024/64 = 16 N-strips
  const int ns = gw & 15;
  const int row0 = mt * 16;
  if (row0 >= M) return;
  const int col0 = ns * 64;
  const int r    = lane & 15;
  const int koff = (lane < 16) ? 0 : 8;
  const int mh   = lane >> 4;

  v8f acc[4] = {v8f{}, v8f{}, v8f{}, v8f{}};
  for (int kc = 0; kc < K; kc += 32) {
    const _Float16* pa = A + (row0 + r) * K + kc + koff;
    v16h a = ldfrag(pa);
#pragma unroll
    for (int g = 0; g < 4; ++g) {
      const _Float16* pb = WT + (col0 + g * 16 + r) * K + kc + koff;
      v16h b = ldfrag(pb);
      acc[g] = __builtin_amdgcn_wmma_f32_16x16x32_f16(
          false, a, false, b, (short)0, acc[g], false, false);
    }
  }
#pragma unroll
  for (int g = 0; g < 4; ++g) {
    const int col = col0 + g * 16 + r;
    const float bv = bias[col];
#pragma unroll
    for (int v = 0; v < 8; ++v) {
      const int m = v + 8 * mh;                  // D layout: lanes>=16 hold M+8
      C[(row0 + m) * N4H + col] = acc[g][v] + bv;
    }
  }
}

// ------------------------- recurrent scan -----------------------------------
// grid = 4: blockIdx -> (dir = b>>1, batch-half = b&1). block = 512 = 16 waves.
// Wave w = ht (hidden tile). Each wave computes the matching i/f/g/o 16x16
// tiles for its hidden columns -> gates & cell state entirely in VGPRs.
__global__ __launch_bounds__(512) void scan_kernel(
    const float* __restrict__ xz_f, const float* __restrict__ xz_b,
    const _Float16* __restrict__ ut_f, const _Float16* __restrict__ ut_b,
    const unsigned char* __restrict__ mask,
    _Float16* __restrict__ o16, float* __restrict__ o32) {
  const int dir = blockIdx.x >> 1;
  const int bh  = blockIdx.x & 1;                // batch half: rows bh*16..+15
  const float*    xz = dir ? xz_b : xz_f;
  const _Float16* UT = dir ? ut_b : ut_f;        // [1024, 256] = U transposed

  const int tid  = threadIdx.x;
  const int ht   = tid >> 5;                     // wave id = hidden tile 0..15
  const int lane = tid & 31;
  const int r    = lane & 15;
  const int koff = (lane < 16) ? 0 : 8;
  const int mh   = lane >> 4;

  __shared__ __align__(16) _Float16 hl[16 * HROW];   // h state (16 rows), f16
  for (int i = tid; i < 16 * HROW; i += 512) hl[i] = (_Float16)0.f;
  __syncthreads();

  v8f creg = v8f{};                              // cell state: VGPRs, all T steps
  float h2v[8];

  for (int s = 0; s < TT; ++s) {
    const int t = dir ? (TT - 1 - s) : s;

    // Prefetch next step's xz gate rows while the GEMM runs.
    if (s + 1 < TT) {
      const int tn = dir ? (TT - 2 - s) : (s + 1);
      const int brow = bh * 16 + mh * 8;
      const long pb = ((long)brow * TT + tn) * (long)N4H + ht * 16 + r;
#pragma unroll
      for (int g = 0; g < 4; ++g) __builtin_prefetch(&xz[pb + g * 256], 0, 3);
    }

    v8f a0 = v8f{}, a1 = v8f{}, a2 = v8f{}, a3 = v8f{};
#pragma unroll
    for (int kc = 0; kc < HH; kc += 32) {
      const _Float16* pa = &hl[r * HROW + kc + koff];
      v16h a = ldfrag(pa);                       // ds_read_b128 x2, conflict-free
      const _Float16* pb0 = UT + (0 * 256 + ht * 16 + r) * HH + kc + koff;
      const _Float16* pb1 = UT + (1 * 256 + ht * 16 + r) * HH + kc + koff;
      const _Float16* pb2 = UT + (2 * 256 + ht * 16 + r) * HH + kc + koff;
      const _Float16* pb3 = UT + (3 * 256 + ht * 16 + r) * HH + kc + koff;
      a0 = __builtin_amdgcn_wmma_f32_16x16x32_f16(false, a, false, ldfrag(pb0), (short)0, a0, false, false);
      a1 = __builtin_amdgcn_wmma_f32_16x16x32_f16(false, a, false, ldfrag(pb1), (short)0, a1, false, false);
      a2 = __builtin_amdgcn_wmma_f32_16x16x32_f16(false, a, false, ldfrag(pb2), (short)0, a2, false, false);
      a3 = __builtin_amdgcn_wmma_f32_16x16x32_f16(false, a, false, ldfrag(pb3), (short)0, a3, false, false);
    }

#pragma unroll
    for (int v = 0; v < 8; ++v) {
      const int m    = mh * 8 + v;               // local batch row 0..15
      const int brow = bh * 16 + m;              // global batch row
      const long base = ((long)brow * TT + t) * (long)N4H + ht * 16 + r;
      const float zi = a0[v] + xz[base];
      const float zf = a1[v] + xz[base + 256];
      const float zg = a2[v] + xz[base + 512];
      const float zo = a3[v] + xz[base + 768];
      const float ig = sigmoidf_(zi);
      const float fg = sigmoidf_(zf);
      const float gg = tanhf(zg);
      const float og = sigmoidf_(zo);
      const float cn = fg * creg[v] + ig * gg;
      const float hn = og * tanhf(cn);
      const bool mk  = mask[(long)brow * TT + t] != 0;
      const float hold = (float)hl[m * HROW + ht * 16 + r];
      h2v[v]  = mk ? hn : hold;
      creg[v] = mk ? cn : creg[v];
    }

    __syncthreads();                             // all reads of old h done
#pragma unroll
    for (int v = 0; v < 8; ++v) {
      const int m    = mh * 8 + v;
      const int brow = bh * 16 + m;
      hl[m * HROW + ht * 16 + r] = (_Float16)h2v[v];
      const long ob = ((long)brow * TT + t) * 512 + dir * 256 + ht * 16 + r;
      if (o16) o16[ob] = (_Float16)h2v[v];       // layer 0 -> f16 for next GEMM
      else     o32[ob] = h2v[v];                 // layer 1 -> final f32 output
    }
    __syncthreads();                             // new h visible before next step
  }
}

// ---------------------------------------------------------------------------

static inline size_t alignup(size_t x) { return (x + 255) & ~(size_t)255; }

extern "C" void kernel_launch(void* const* d_in, const int* in_sizes, int n_in,
                              void* d_out, int out_size, void* d_ws, size_t ws_size,
                              hipStream_t stream) {
  (void)in_sizes; (void)n_in; (void)out_size; (void)ws_size;
  const float* x    = (const float*)d_in[0];
  const unsigned char* mask = (const unsigned char*)d_in[1];   // jnp bool = 1 byte
  const float* W_f0 = (const float*)d_in[2];
  const float* U_f0 = (const float*)d_in[3];
  const float* b_f0 = (const float*)d_in[4];
  const float* W_b0 = (const float*)d_in[5];
  const float* U_b0 = (const float*)d_in[6];
  const float* b_b0 = (const float*)d_in[7];
  const float* W_f1 = (const float*)d_in[8];
  const float* U_f1 = (const float*)d_in[9];
  const float* b_f1 = (const float*)d_in[10];
  const float* W_b1 = (const float*)d_in[11];
  const float* U_b1 = (const float*)d_in[12];
  const float* b_b1 = (const float*)d_in[13];
  float* out = (float*)d_out;

  const int M = BB * TT;                       // 65536
  char* ws = (char*)d_ws;
  size_t off = 0;
  _Float16* xh   = (_Float16*)(ws + off); off = alignup(off + (size_t)M * EE * 2);
  _Float16* h1h  = (_Float16*)(ws + off); off = alignup(off + (size_t)M * 512 * 2);
  float*    xzf  = (float*)(ws + off);    off = alignup(off + (size_t)M * N4H * 4);
  float*    xzb  = (float*)(ws + off);    off = alignup(off + (size_t)M * N4H * 4);
  _Float16* wtf0 = (_Float16*)(ws + off); off = alignup(off + (size_t)N4H * EE * 2);
  _Float16* wtb0 = (_Float16*)(ws + off); off = alignup(off + (size_t)N4H * EE * 2);
  _Float16* utf0 = (_Float16*)(ws + off); off = alignup(off + (size_t)N4H * HH * 2);
  _Float16* utb0 = (_Float16*)(ws + off); off = alignup(off + (size_t)N4H * HH * 2);
  _Float16* wtf1 = (_Float16*)(ws + off); off = alignup(off + (size_t)N4H * 512 * 2);
  _Float16* wtb1 = (_Float16*)(ws + off); off = alignup(off + (size_t)N4H * 512 * 2);
  _Float16* utf1 = (_Float16*)(ws + off); off = alignup(off + (size_t)N4H * HH * 2);
  _Float16* utb1 = (_Float16*)(ws + off); off = alignup(off + (size_t)N4H * HH * 2);

  // --- weight prep (tiny) ---
  {
    int n = M * EE;
    cvt_f16_kernel<<<(n + 255) / 256, 256, 0, stream>>>(x, xh, n);
    auto tp = [&](const float* s, _Float16* dd, int K, int N) {
      int e = K * N;
      transpose_cvt_kernel<<<(e + 255) / 256, 256, 0, stream>>>(s, dd, K, N);
    };
    tp(W_f0, wtf0, EE, N4H);  tp(W_b0, wtb0, EE, N4H);
    tp(U_f0, utf0, HH, N4H);  tp(U_b0, utb0, HH, N4H);
    tp(W_f1, wtf1, 512, N4H); tp(W_b1, wtb1, 512, N4H);
    tp(U_f1, utf1, HH, N4H);  tp(U_b1, utb1, HH, N4H);
  }

  const int projBlocks = (M / 16) * 16 * 32 / 256;   // waves = (M/16)*16 strips

  // --- layer 0 ---
  proj_kernel<<<projBlocks, 256, 0, stream>>>(xh, wtf0, b_f0, xzf, M, EE);
  proj_kernel<<<projBlocks, 256, 0, stream>>>(xh, wtb0, b_b0, xzb, M, EE);
  scan_kernel<<<4, 512, 0, stream>>>(xzf, xzb, utf0, utb0, mask, h1h, nullptr);

  // --- layer 1 (XZ buffers reused) ---
  proj_kernel<<<projBlocks, 256, 0, stream>>>(h1h, wtf1, b_f1, xzf, M, 512);
  proj_kernel<<<projBlocks, 256, 0, stream>>>(h1h, wtb1, b_b1, xzb, M, 512);
  scan_kernel<<<4, 512, 0, stream>>>(xzf, xzb, utf1, utb1, mask, nullptr, out);
}
